// JKNet_30322469110222
// MI455X (gfx1250) — matrix operand order repaired and checked by reading, not compiled, not run on hardware
//
#include <hip/hip_runtime.h>
#include <hip/hip_bf16.h>

// ---------------------------------------------------------------------------
// JKNet inference for MI455X (gfx1250, wave32, WMMA).
// Pipeline: bf16-convert -> [WMMA GEMM -> SpMM scatter (f32 atomics) -> ReLU+cvt] x6
//           -> WMMA GEMM (jumping-knowledge FC) -> bias + log_softmax.
// ---------------------------------------------------------------------------

typedef __attribute__((ext_vector_type(16))) __bf16 v16bf;
typedef __attribute__((ext_vector_type(8)))  __bf16 v8bf;
typedef __attribute__((ext_vector_type(4)))  __bf16 v4bf;
typedef __attribute__((ext_vector_type(8)))  float  v8f;

#define NFEAT  512
#define NHID   128
#define NLAYER 6
#define NCLASS 64
#define NCAT   (NHID * NLAYER)   // 768

// ---------------------------------------------------------------------------
// f32 -> bf16 elementwise (vectorized x4). n must be a multiple of 4.
// ---------------------------------------------------------------------------
__global__ __launch_bounds__(256)
void cvt_f32_bf16(const float* __restrict__ in, __bf16* __restrict__ out, long n) {
    long i = ((long)blockIdx.x * 256 + threadIdx.x) * 4;
    if (i >= n) return;
    float4 v = *(const float4*)(in + i);
    v4bf o;
    o[0] = (__bf16)v.x; o[1] = (__bf16)v.y; o[2] = (__bf16)v.z; o[3] = (__bf16)v.w;
    *(v4bf*)(out + i) = o;
}

// ---------------------------------------------------------------------------
// Transpose + convert:  W[R][C] (f32, row-major) -> Wt[C][R] (bf16, row-major).
// Weights are tiny; scalar transpose is fine.
// ---------------------------------------------------------------------------
__global__ __launch_bounds__(256)
void transpose_cvt(const float* __restrict__ W, __bf16* __restrict__ Wt, int R, int C) {
    int i = blockIdx.x * 256 + threadIdx.x;
    if (i >= R * C) return;
    int r = i / C;
    int c = i - r * C;
    Wt[(size_t)c * R + r] = (__bf16)W[i];
}

// ---------------------------------------------------------------------------
// WMMA GEMM:  C[M x (NT*16)] = A[M x K] (bf16, lda) * Bt[(NT*16) x K]^T (bf16, ldb=K)
// One wave computes a full 16-row x (NT*16)-col stripe: A fragment loaded once
// per k-step and reused across all NT B tiles (exact A traffic).
//
// Fragment layouts used here (16-bit, wave32):
//   A 16x32: lane&15 = row M; half=lane>>4: K = half*16 .. +15 (contiguous)
//   B 32x16: lane&15 = col N; half=lane>>4: K = half*16 .. +15 (contiguous)
//   C 16x16 f32: lane&15 = col N; vgpr r = row half*8 + r
// ---------------------------------------------------------------------------
template <int NT>
__global__ __launch_bounds__(256)
void gemm_bf16_wmma(const __bf16* __restrict__ A, int lda,
                    const __bf16* __restrict__ Bt, int ldb,
                    float* __restrict__ C, int ldc,
                    int K, int n_mtiles) {
    const int lane  = threadIdx.x & 31;
    const int wave  = threadIdx.x >> 5;
    const int mtile = blockIdx.x * 8 + wave;     // wave-uniform -> EXEC stays full
    if (mtile >= n_mtiles) return;

    const int m    = lane & 15;
    const int half = lane >> 4;

    const __bf16* arow = A + (size_t)(mtile * 16 + m) * lda;

    v8f acc[NT];
#pragma unroll
    for (int t = 0; t < NT; ++t) acc[t] = {};

    for (int k0 = 0; k0 < K; k0 += 32) {
        // prefetch the next A chunk for this row (global_prefetch_b8)
        if (k0 + 32 < K) __builtin_prefetch(arow + k0 + 32, 0, 0);

        v16bf a  = *(const v16bf*)(arow + k0 + half * 16);
#pragma unroll
        for (int t = 0; t < NT; ++t) {
            const __bf16* brow = Bt + (size_t)(t * 16 + m) * ldb + k0 + half * 16;
            v16bf b = *(const v16bf*)brow;
            acc[t] = __builtin_amdgcn_wmma_f32_16x16x32_bf16(
                         false, a, false, b, (short)0, acc[t], false, false);
        }
    }

    float* crow = C + (size_t)(mtile * 16 + half * 8) * ldc + m;
#pragma unroll
    for (int t = 0; t < NT; ++t) {
#pragma unroll
        for (int r = 0; r < 8; ++r) {
            crow[(size_t)r * ldc + t * 16] = acc[t][r];
        }
    }
}

// ---------------------------------------------------------------------------
// SpMM scatter: one wave per edge. Gather source row (128 f32, L2-resident),
// scale by edge weight, scatter-add into dest row with global_atomic_add_f32.
// ---------------------------------------------------------------------------
__global__ __launch_bounds__(256)
void spmm_scatter(const int* __restrict__ rows, const int* __restrict__ cols,
                  const float* __restrict__ vals, const float* __restrict__ z,
                  float* __restrict__ acc, int nedges) {
    int gid  = blockIdx.x * 256 + threadIdx.x;
    int e    = gid >> 5;
    int lane = gid & 31;
    if (e >= nedges) return;

    int   r = rows[e];
    int   c = cols[e];
    float v = vals[e];

    float4 d = ((const float4*)(z + (size_t)c * NHID))[lane];
    float* dst = acc + (size_t)r * NHID + lane * 4;
    atomicAdd(dst + 0, v * d.x);
    atomicAdd(dst + 1, v * d.y);
    atomicAdd(dst + 2, v * d.z);
    atomicAdd(dst + 3, v * d.w);
}

// ---------------------------------------------------------------------------
// ReLU + bf16 convert, writing into a 128-wide slice of the 768-wide concat
// buffer (ld_out = 768). One thread per 4 features.
// ---------------------------------------------------------------------------
__global__ __launch_bounds__(256)
void relu_cvt(const float* __restrict__ acc, __bf16* __restrict__ out,
              int ld_out, int n_nodes) {
    int gid = blockIdx.x * 256 + threadIdx.x;
    if (gid >= n_nodes * 32) return;
    int node = gid >> 5;
    int q    = gid & 31;
    float4 d = ((const float4*)(acc + (size_t)node * NHID))[q];
    v4bf o;
    o[0] = (__bf16)fmaxf(d.x, 0.f);
    o[1] = (__bf16)fmaxf(d.y, 0.f);
    o[2] = (__bf16)fmaxf(d.z, 0.f);
    o[3] = (__bf16)fmaxf(d.w, 0.f);
    *(v4bf*)(out + (size_t)node * ld_out + q * 4) = o;
}

// ---------------------------------------------------------------------------
// bias add + log_softmax over 64 classes. One wave per row, 2 classes/lane,
// wave32 shuffle reductions.
// ---------------------------------------------------------------------------
__global__ __launch_bounds__(256)
void bias_logsoftmax(const float* __restrict__ logits, const float* __restrict__ bfc,
                     float* __restrict__ out, int n_nodes) {
    int gid  = blockIdx.x * 256 + threadIdx.x;
    int row  = gid >> 5;
    int lane = gid & 31;
    if (row >= n_nodes) return;

    float2 l = ((const float2*)(logits + (size_t)row * NCLASS))[lane];
    float2 b = ((const float2*)bfc)[lane];
    float x0 = l.x + b.x;
    float x1 = l.y + b.y;

    float m = fmaxf(x0, x1);
#pragma unroll
    for (int s = 16; s > 0; s >>= 1) m = fmaxf(m, __shfl_xor(m, s, 32));

    float sum = expf(x0 - m) + expf(x1 - m);
#pragma unroll
    for (int s = 16; s > 0; s >>= 1) sum += __shfl_xor(sum, s, 32);

    float lse = m + logf(sum);
    float2 o = make_float2(x0 - lse, x1 - lse);
    ((float2*)(out + (size_t)row * NCLASS))[lane] = o;
}

// ---------------------------------------------------------------------------
// Host-side orchestration.
// ---------------------------------------------------------------------------
extern "C" void kernel_launch(void* const* d_in, const int* in_sizes, int n_in,
                              void* d_out, int out_size, void* d_ws, size_t ws_size,
                              hipStream_t stream) {
    const float* x    = (const float*)d_in[0];
    const int*   rows = (const int*)d_in[1];
    const int*   cols = (const int*)d_in[2];
    const float* vals = (const float*)d_in[3];
    const float* W0   = (const float*)d_in[4];
    const float* Wh   = (const float*)d_in[5];
    const float* Wfc  = (const float*)d_in[6];
    const float* bfc  = (const float*)d_in[7];

    const int NN = in_sizes[0] / NFEAT;   // 100000 nodes
    const int NE = in_sizes[1];           // 1.6M edges

    // ---- carve workspace (256B aligned regions) ----
    char* p = (char*)d_ws;
    auto carve = [&](size_t bytes) -> char* {
        char* r = p;
        p += (bytes + 255) & ~(size_t)255;
        return r;
    };
    __bf16* x16    = (__bf16*)carve((size_t)NN * NFEAT * 2);     // 102.4 MB
    __bf16* hidden = (__bf16*)carve((size_t)NN * NCAT * 2);      // 153.6 MB
    float*  z      = (float*) carve((size_t)NN * NHID * 4);      //  51.2 MB (also logits)
    float*  accb   = (float*) carve((size_t)NN * NHID * 4);      //  51.2 MB
    __bf16* W0t    = (__bf16*)carve((size_t)NFEAT * NHID * 2);
    __bf16* Wht    = (__bf16*)carve((size_t)(NLAYER - 1) * NHID * NHID * 2);
    __bf16* Wfct   = (__bf16*)carve((size_t)NCAT * NCLASS * 2);

    // ---- precision conversion / weight transposes ----
    {
        long n = (long)NN * NFEAT;
        cvt_f32_bf16<<<(int)((n / 4 + 255) / 256), 256, 0, stream>>>(x, x16, n);
    }
    transpose_cvt<<<(NFEAT * NHID + 255) / 256, 256, 0, stream>>>(W0, W0t, NFEAT, NHID);
    for (int i = 0; i < NLAYER - 1; ++i) {
        transpose_cvt<<<(NHID * NHID + 255) / 256, 256, 0, stream>>>(
            Wh + (size_t)i * NHID * NHID, Wht + (size_t)i * NHID * NHID, NHID, NHID);
    }
    transpose_cvt<<<(NCAT * NCLASS + 255) / 256, 256, 0, stream>>>(Wfc, Wfct, NCAT, NCLASS);

    const int n_mtiles = NN / 16;                  // 6250
    const int gblocks  = (n_mtiles + 7) / 8;       // 8 waves (m-tiles) per block
    const int sblocks  = (NE + 7) / 8;             // 8 edges (waves) per block
    const int rblocks  = (NN * 32 + 255) / 256;

    // ---- layer 0: relu(spmm(A, x @ W0)) ----
    gemm_bf16_wmma<NHID / 16><<<gblocks, 256, 0, stream>>>(
        x16, NFEAT, W0t, NFEAT, z, NHID, NFEAT, n_mtiles);
    hipMemsetAsync(accb, 0, (size_t)NN * NHID * 4, stream);
    spmm_scatter<<<sblocks, 256, 0, stream>>>(rows, cols, vals, z, accb, NE);
    relu_cvt<<<rblocks, 256, 0, stream>>>(accb, hidden, NCAT, NN);

    // ---- layers 1..5: relu(spmm(A, h @ Wh[l-1])) ----
    for (int l = 1; l < NLAYER; ++l) {
        gemm_bf16_wmma<NHID / 16><<<gblocks, 256, 0, stream>>>(
            hidden + (size_t)(l - 1) * NHID, NCAT,
            Wht + (size_t)(l - 1) * NHID * NHID, NHID,
            z, NHID, NHID, n_mtiles);
        hipMemsetAsync(accb, 0, (size_t)NN * NHID * 4, stream);
        spmm_scatter<<<sblocks, 256, 0, stream>>>(rows, cols, vals, z, accb, NE);
        relu_cvt<<<rblocks, 256, 0, stream>>>(accb, hidden + (size_t)l * NHID, NCAT, NN);
    }

    // ---- jumping-knowledge FC: concat(hiddens) @ Wfc + bfc, then log_softmax ----
    gemm_bf16_wmma<NCLASS / 16><<<gblocks, 256, 0, stream>>>(
        hidden, NCAT, Wfct, NCAT, z /* logits */, NCLASS, NCAT, n_mtiles);
    bias_logsoftmax<<<rblocks, 256, 0, stream>>>(z, bfc, (float*)d_out, NN);
}
